// MultiHeadAttention_4621384810513
// MI455X (gfx1250) — compile-verified, matrix operand order checked by reading, not run on hardware
//
#include <hip/hip_runtime.h>
#include <hip/hip_bf16.h>
#include <stdint.h>

// ---------------------------------------------------------------------------
// MHA block for MI455X (gfx1250, wave32, WMMA).
// B=4, L=2048, D_MODEL=512, H=8, Dk=Dv=64.
// bf16 WMMA (v_wmma_f32_16x16x32_bf16) with f32 accumulation everywhere;
// softmax / bias / residual / LayerNorm in f32.
// ---------------------------------------------------------------------------

#define B_  4
#define L_  2048
#define DM  512
#define NH  8
#define DK  64
#define PITCH 2052            // f32 LDS row pitch for the 16 x 2048 score tile

typedef __attribute__((ext_vector_type(16))) __bf16        v16bf;
typedef __attribute__((ext_vector_type(8)))  float         v8f;
typedef __attribute__((ext_vector_type(4)))  float         f32x4;
typedef __attribute__((ext_vector_type(4)))  unsigned int  u32x4;

union Frag {                  // 16 bf16 = one WMMA A/B operand (8 VGPRs)
  v16bf        v;
  unsigned short u[16];
  u32x4        q4[2];
};

__device__ __forceinline__ unsigned short f2bf(float f) {
  return __builtin_bit_cast(unsigned short, (__bf16)f);   // native v_cvt path
}

__device__ __forceinline__ v8f wmma_bf16(const Frag& a, const Frag& b, v8f c) {
  return __builtin_amdgcn_wmma_f32_16x16x32_bf16(false, a.v, false, b.v,
                                                 (short)0, c, false, false);
}

// A/B fragment from row-major bf16 memory. rowptr = row (lane&15) of the tile.
// k element j maps to kbase + (j<8 ? j : j+8) + (lane&16 ? 8 : 0).
__device__ __forceinline__ Frag load_bf16_frag(const unsigned short* rowptr,
                                               int kbase, int lane) {
  Frag f;
  const int off = kbase + ((lane & 16) ? 8 : 0);
  const u32x4* p = (const u32x4*)(rowptr + off);   // 16B-aligned
  f.q4[0] = p[0];        // k = off .. off+7
  f.q4[1] = p[2];        // k = off+16 .. off+23
  return f;
}

// Same mapping, but source is f32 (weights / activations), converted bf16.
__device__ __forceinline__ Frag load_f32_frag(const float* rowptr,
                                              int kbase, int lane) {
  Frag f;
  const int off = kbase + ((lane & 16) ? 8 : 0);
  const f32x4* p = (const f32x4*)(rowptr + off);
  f32x4 a0 = p[0], a1 = p[1];   // k off..off+7
  f32x4 b0 = p[4], b1 = p[5];   // k off+16..off+23
#pragma unroll
  for (int i = 0; i < 4; ++i) {
    f.v[i]      = (__bf16)a0[i];
    f.v[4 + i]  = (__bf16)a1[i];
    f.v[8 + i]  = (__bf16)b0[i];
    f.v[12 + i] = (__bf16)b1[i];
  }
  return f;
}

__device__ __forceinline__ float wave_max(float v) {
#pragma unroll
  for (int m = 16; m >= 1; m >>= 1) v = fmaxf(v, __shfl_xor(v, m, 32));
  return v;
}
__device__ __forceinline__ float wave_sum(float v) {
#pragma unroll
  for (int m = 16; m >= 1; m >>= 1) v += __shfl_xor(v, m, 32);
  return v;
}

// ---------------------------------------------------------------------------
// Kernel 1: fused Q/K/V projection.  grid=(BL/16, 3), block=256 (8 waves).
// Wave w computes 64 output columns (4 N-tiles) so the activation A-fragment
// is loaded once per wave per K-step.
// Q,K stored bf16 [B,H,L,64]; V stored transposed bf16 [B,H,64,L].
// ---------------------------------------------------------------------------
__global__ __launch_bounds__(256)
void qkv_proj_kernel(const float* __restrict__ xq, const float* __restrict__ xk,
                     const float* __restrict__ xv,
                     const float* __restrict__ wq, const float* __restrict__ bq,
                     const float* __restrict__ wk, const float* __restrict__ bk,
                     const float* __restrict__ wv, const float* __restrict__ bv,
                     unsigned short* __restrict__ Qh,
                     unsigned short* __restrict__ Kh,
                     unsigned short* __restrict__ Vt) {
  const int lane  = threadIdx.x & 31;
  const int wave  = threadIdx.x >> 5;
  const int which = blockIdx.y;              // 0=Q 1=K 2=V
  const int row0  = blockIdx.x * 16;         // row in [B*L]
  const int n     = lane & 15;
  const int hi8   = (lane & 16) ? 8 : 0;

  const float* x    = (which == 0) ? xq : (which == 1) ? xk : xv;
  const float* w    = (which == 0) ? wq : (which == 1) ? wk : wv;
  const float* bias = (which == 0) ? bq : (which == 1) ? bk : bv;

  const float* arow = x + (size_t)(row0 + n) * DM;

  v8f acc[4] = {{}, {}, {}, {}};
  for (int kk = 0; kk < DM; kk += 32) {
    Frag A = load_f32_frag(arow, kk, lane);
#pragma unroll
    for (int tt = 0; tt < 4; ++tt) {
      const int ocb = wave * 64 + tt * 16;   // output column base (0..511)
      Frag Bf = load_f32_frag(w + (size_t)(ocb + n) * DM, kk, lane);
      acc[tt] = wmma_bf16(A, Bf, acc[tt]);
    }
  }

#pragma unroll
  for (int tt = 0; tt < 4; ++tt) {
    const int ocol = wave * 64 + tt * 16 + n;  // 0..511
    const int h    = ocol >> 6;
    const int col  = ocol & 63;
    const float bs = bias[ocol];
#pragma unroll
    for (int r = 0; r < 8; ++r) {
      const int m    = r + hi8;
      const int rowg = row0 + m;
      const int bb   = rowg >> 11;             // /L_
      const int l    = rowg & (L_ - 1);
      const unsigned short bvv = f2bf(acc[tt][r] + bs);
      if (which == 2)
        Vt[((size_t)(bb * NH + h) * DK + col) * L_ + l] = bvv;
      else if (which == 0)
        Qh[((size_t)(bb * NH + h) * L_ + l) * DK + col] = bvv;
      else
        Kh[((size_t)(bb * NH + h) * L_ + l) * DK + col] = bvv;
    }
  }
}

// ---------------------------------------------------------------------------
// Kernel 2: scores + mask + softmax + attn write + P@V.
// grid=(L/16, NH, B), block=256 (8 waves), dynamic LDS:
//   sc   : 16 x PITCH f32 score/prob tile; after softmax the low half of each
//          row additionally holds the row's 2048 probs as bf16 (in-place)
//   opart: 4 x 16 x 16 f32 cross-wave P@V partials
// ---------------------------------------------------------------------------
__global__ __launch_bounds__(256)
void attn_kernel(const unsigned short* __restrict__ Qh,
                 const unsigned short* __restrict__ Kh,
                 const unsigned short* __restrict__ Vt,
                 const unsigned char*  __restrict__ mask,
                 float* __restrict__ attnOut,            // [H,B,L,L]
                 unsigned short* __restrict__ Hout) {    // [B,L,H*64] bf16
  extern __shared__ float smem[];
  float* sc    = smem;                  // 16 * PITCH
  float* opart = smem + 16 * PITCH;     // 4 * 256

  const int lane = threadIdx.x & 31;
  const int wave = threadIdx.x >> 5;
  const int q0   = blockIdx.x * 16;
  const int h    = blockIdx.y;
  const int bz   = blockIdx.z;
  const int n    = lane & 15;
  const int hi8  = (lane & 16) ? 8 : 0;

  const unsigned short* qbase = Qh + (size_t)(bz * NH + h) * L_ * DK;
  const unsigned short* kbase = Kh + (size_t)(bz * NH + h) * L_ * DK;
  const unsigned short* vbase = Vt + (size_t)(bz * NH + h) * DK * L_;
  const unsigned char*  mbase = mask + (size_t)bz * L_ * L_;

  // Q operand for this query tile (shared by all score tiles of this wave)
  const unsigned short* qrow = qbase + (size_t)(q0 + n) * DK;
  Frag Aq0 = load_bf16_frag(qrow, 0,  lane);
  Frag Aq1 = load_bf16_frag(qrow, 32, lane);

  // ---- scores: each wave owns key tiles t = wave, wave+8, ... (16 tiles) ----
  for (int t = wave; t < L_ / 16; t += 8) {
    const unsigned short* krow = kbase + (size_t)(t * 16 + n) * DK;
    if (t + 8 < L_ / 16)
      __builtin_prefetch(kbase + (size_t)((t + 8) * 16 + n) * DK, 0, 1);
    Frag Bk0 = load_bf16_frag(krow, 0,  lane);
    Frag Bk1 = load_bf16_frag(krow, 32, lane);
    v8f c = {};
    c = wmma_bf16(Aq0, Bk0, c);
    c = wmma_bf16(Aq1, Bk1, c);
#pragma unroll
    for (int r = 0; r < 8; ++r) {
      const int m   = r + hi8;
      const int col = t * 16 + n;
      const unsigned char mk = mbase[(size_t)(q0 + m) * L_ + col];
      sc[m * PITCH + col] = mk ? -__builtin_inff() : c[r] * 0.125f;
    }
  }
  __syncthreads();

  // ---- exact softmax: wave w owns rows 2w, 2w+1 (full 2048-wide rows) ----
#pragma unroll
  for (int rr = 0; rr < 2; ++rr) {
    const int m = wave * 2 + rr;
    float mx = -__builtin_inff();
    for (int i = lane; i < L_; i += 32) mx = fmaxf(mx, sc[m * PITCH + i]);
    mx = wave_max(mx);
    float s = 0.f;
    for (int i = lane; i < L_; i += 32) {
      const float e = __expf(sc[m * PITCH + i] - mx);
      sc[m * PITCH + i] = e;
      s += e;
    }
    s = wave_sum(s);
    const float inv = 1.f / s;
    float* arow_out = attnOut + ((size_t)(h * B_ + bz) * L_ + (q0 + m)) * L_;
    unsigned short* prow16 = (unsigned short*)(sc + (size_t)m * PITCH);
    // In-place bf16 compaction: u16 write at index i clobbers f32 element i/2,
    // which is already consumed (same-step overlap covered by per-wave
    // in-order LDS execution; this wave exclusively owns rows 2w,2w+1).
    for (int i = lane; i < L_; i += 32) {
      const float p = sc[m * PITCH + i] * inv;
      arow_out[i] = p;                       // coalesced f32 write (536MB total)
      prow16[i]   = f2bf(p);                 // bf16 probs for the P@V stage
    }
  }
  __syncthreads();

  // ---- P @ V : wave = (khalf, ntile); each accumulates 16x16 over 1024 keys ----
  const int ntile = wave & 3;                // output d-column tile (4 x 16 = 64)
  const int khalf = wave >> 2;               // key range half
  const unsigned short* vrow   = vbase + (size_t)(ntile * 16 + n) * L_;
  const unsigned short* prow16 = (const unsigned short*)(sc + (size_t)n * PITCH);

  v8f acc = {};
  for (int i = 0; i < 32; ++i) {
    const int kc = (khalf * 32 + i) * 32;    // key base for this 16x16x32 step
    if (i + 1 < 32)
      __builtin_prefetch(vrow + kc + 32, 0, 1);
    Frag Ap = load_bf16_frag(prow16, kc, lane);   // 2x ds_load_b128
    Frag Bv = load_bf16_frag(vrow,   kc, lane);
    acc = wmma_bf16(Ap, Bv, acc);
  }

  if (khalf == 1) {
#pragma unroll
    for (int r = 0; r < 8; ++r)
      opart[ntile * 256 + (r + hi8) * 16 + n] = acc[r];
  }
  __syncthreads();
  if (khalf == 0) {
#pragma unroll
    for (int r = 0; r < 8; ++r) {
      const int m = r + hi8;
      const float o = acc[r] + opart[ntile * 256 + m * 16 + n];
      Hout[((size_t)(bz * L_ + q0 + m) * DM) + h * 64 + ntile * 16 + n] = f2bf(o);
    }
  }
}

// ---------------------------------------------------------------------------
// Kernel 3: FC + bias + residual + LayerNorm.  grid=(BL/16), block=256.
// ---------------------------------------------------------------------------
#define XP 516
__global__ __launch_bounds__(256)
void fc_ln_kernel(const unsigned short* __restrict__ Hout,
                  const float* __restrict__ fcw, const float* __restrict__ fcb,
                  const float* __restrict__ qin,
                  const float* __restrict__ lng, const float* __restrict__ lnb,
                  float* __restrict__ out) {
  __shared__ float X[16 * XP];

  const int lane = threadIdx.x & 31;
  const int wave = threadIdx.x >> 5;
  const int row0 = blockIdx.x * 16;
  const int n    = lane & 15;
  const int hi8  = (lane & 16) ? 8 : 0;

  const unsigned short* arow = Hout + (size_t)(row0 + n) * DM;

  v8f acc[4] = {{}, {}, {}, {}};
  for (int kk = 0; kk < DM; kk += 32) {
    Frag A = load_bf16_frag(arow, kk, lane);
#pragma unroll
    for (int tt = 0; tt < 4; ++tt) {
      const int colb = wave * 64 + tt * 16;
      Frag Bf = load_f32_frag(fcw + (size_t)(colb + n) * DM, kk, lane);
      acc[tt] = wmma_bf16(A, Bf, acc[tt]);
    }
  }

#pragma unroll
  for (int tt = 0; tt < 4; ++tt) {
    const int col = wave * 64 + tt * 16 + n;
    const float bs = fcb[col];
#pragma unroll
    for (int r = 0; r < 8; ++r) {
      const int m = r + hi8;
      X[m * XP + col] = acc[tt][r] + bs + qin[(size_t)(row0 + m) * DM + col];
    }
  }
  __syncthreads();

  // LayerNorm: wave w owns rows 2w, 2w+1
#pragma unroll
  for (int rr = 0; rr < 2; ++rr) {
    const int m = wave * 2 + rr;
    float s = 0.f, s2 = 0.f;
    for (int i = lane; i < DM; i += 32) {
      const float x = X[m * XP + i];
      s += x; s2 += x * x;
    }
    s = wave_sum(s); s2 = wave_sum(s2);
    const float mu  = s * (1.0f / DM);
    const float var = s2 * (1.0f / DM) - mu * mu;
    const float rs  = rsqrtf(var + 1e-5f);
    float* orow = out + (size_t)(row0 + m) * DM;
    for (int i = lane; i < DM; i += 32)
      orow[i] = (X[m * XP + i] - mu) * rs * lng[i] + lnb[i];
  }
}

// ---------------------------------------------------------------------------
extern "C" void kernel_launch(void* const* d_in, const int* in_sizes, int n_in,
                              void* d_out, int out_size, void* d_ws, size_t ws_size,
                              hipStream_t stream) {
  const float* q    = (const float*)d_in[0];
  const float* k    = (const float*)d_in[1];
  const float* v    = (const float*)d_in[2];
  const unsigned char* mask = (const unsigned char*)d_in[3];
  const float* wq_w = (const float*)d_in[4];
  const float* wq_b = (const float*)d_in[5];
  const float* wk_w = (const float*)d_in[6];
  const float* wk_b = (const float*)d_in[7];
  const float* wv_w = (const float*)d_in[8];
  const float* wv_b = (const float*)d_in[9];
  const float* fc_w = (const float*)d_in[10];
  const float* fc_b = (const float*)d_in[11];
  const float* ln_g = (const float*)d_in[12];
  const float* ln_b = (const float*)d_in[13];

  float* out  = (float*)d_out;                          // [B,L,512]
  float* attn = out + (size_t)B_ * L_ * DM;             // [H*B,L,L]

  const size_t NPH = (size_t)B_ * NH * L_ * DK;         // 4,194,304
  unsigned short* Qh   = (unsigned short*)d_ws;
  unsigned short* Kh   = Qh + NPH;
  unsigned short* Vt   = Kh + NPH;
  unsigned short* Hout = Vt + NPH;                      // [B,L,512] bf16

  qkv_proj_kernel<<<dim3((B_ * L_) / 16, 3), 256, 0, stream>>>(
      q, k, v, wq_w, wq_b, wk_w, wk_b, wv_w, wv_b, Qh, Kh, Vt);

  const size_t smem = (size_t)(16 * PITCH + 4 * 256) * sizeof(float); // ~135 KB
  attn_kernel<<<dim3(L_ / 16, NH, B_), 256, smem, stream>>>(
      Qh, Kh, Vt, mask, attn, Hout);

  fc_ln_kernel<<<dim3((B_ * L_) / 16), 256, 0, stream>>>(
      Hout, fc_w, fc_b, q, ln_g, ln_b, out);
}
